// CoAttnLayer_82884278878794
// MI455X (gfx1250) — matrix operand order repaired and checked by reading, not compiled, not run on hardware
//
#include <hip/hip_runtime.h>
#include <hip/hip_bf16.h>

typedef __attribute__((ext_vector_type(16))) __bf16 v16bf;
typedef __attribute__((ext_vector_type(8)))  __bf16 v8bf;
typedef __attribute__((ext_vector_type(4)))  __bf16 v4bf;
typedef __attribute__((ext_vector_type(8)))  float  v8f;
typedef __attribute__((ext_vector_type(4)))  unsigned int u32x4;
typedef __attribute__((ext_vector_type(8)))  int  i32x8;
typedef __attribute__((ext_vector_type(4)))  int  i32x4;

#define NEG_INF (-__builtin_inff())

// ---- TDM availability (evaluated per compile pass; host pass parses fallback) ----
#if defined(__has_builtin)
#  if __has_builtin(__builtin_amdgcn_tensor_load_to_lds) && __has_builtin(__builtin_amdgcn_s_wait_tensorcnt)
#    define HAVE_TDM 1
#  endif
#endif
#ifndef HAVE_TDM
#  define HAVE_TDM 0
#endif
#if defined(__has_include)
#  if __has_include(<hip/amd_detail/amd_gfx1250_TDM.h>)
#    define TDM_6ARG 1
#  endif
#endif
#ifndef TDM_6ARG
#  define TDM_6ARG 0
#endif

// ---------------- f32 -> bf16 conversion (vectorized x4) ----------------
__global__ __launch_bounds__(256) void cvt_bf16_kernel(const float* __restrict__ in,
                                                       __bf16* __restrict__ out,
                                                       long long n4) {
    long long i = (long long)blockIdx.x * 256 + threadIdx.x;
    if (i >= n4) return;
    float4 v = ((const float4*)in)[i];
    v4bf o;
    o[0] = (__bf16)v.x; o[1] = (__bf16)v.y; o[2] = (__bf16)v.z; o[3] = (__bf16)v.w;
    ((v4bf*)out)[i] = o;
}

// ---------------- WMMA fragment loaders (ISA 7.12.2 layouts) ----------------
// LDS tiles are [128 rows][40 bf16] (row stride 40 => 80B, 16B-aligned sublocs).
#define LDS_STRIDE 40
#define TILE_BYTES (128 * LDS_STRIDE * 2)

// A fragment 16x32 bf16: lane m = lane&15, h = lane>>4.
__device__ __forceinline__ v16bf load_fragA(const __bf16* tile, int row0, int lane) {
    const int m = lane & 15, h = lane >> 4;
    const __bf16* p = tile + (row0 + m) * LDS_STRIDE;
    v8bf lo = *(const v8bf*)(p + 8 * h);
    v8bf hi = *(const v8bf*)(p + 16 + 8 * h);
    v16bf r;
#pragma unroll
    for (int i = 0; i < 8; ++i) { r[i] = lo[i]; r[i + 8] = hi[i]; }
    return r;
}

// B fragment 32x16 bf16 (tile stored [n][k]): lane n = lane&15, h = lane>>4.
__device__ __forceinline__ v16bf load_fragB(const __bf16* tile, int row0, int lane) {
    const int n = lane & 15, h = lane >> 4;
    const __bf16* p = tile + (row0 + n) * LDS_STRIDE;
    v8bf lo = *(const v8bf*)(p + 8 * h);
    v8bf hi = *(const v8bf*)(p + 16 + 8 * h);
    v16bf r;
#pragma unroll
    for (int i = 0; i < 8; ++i) { r[i] = lo[i]; r[i + 8] = hi[i]; }
    return r;
}

// 8 WMMAs for one 32-wide K step; wave tile 32x64 (2x4 accumulators)
__device__ __forceinline__ void mma_step(const __bf16* Asb, const __bf16* Bsb,
                                         int wm, int wn, int lane, v8f (&acc)[2][4]) {
    v16bf af0 = load_fragA(Asb, wm + 0, lane);
    v16bf af1 = load_fragA(Asb, wm + 16, lane);
#pragma unroll
    for (int ni = 0; ni < 4; ++ni) {
        v16bf bf = load_fragB(Bsb, wn + ni * 16, lane);
        acc[0][ni] = __builtin_amdgcn_wmma_f32_16x16x32_bf16(
            false, af0, false, bf, (short)0, acc[0][ni], false, false);
        acc[1][ni] = __builtin_amdgcn_wmma_f32_16x16x32_bf16(
            false, af1, false, bf, (short)0, acc[1][ni], false, false);
    }
}

template <typename OutT>
__device__ __forceinline__ void store_tile(OutT* __restrict__ C, int N, int m0, int n0,
                                           int wm, int wn, int lane, v8f (&acc)[2][4]) {
    const int nl = lane & 15, hl = lane >> 4;
#pragma unroll
    for (int mi = 0; mi < 2; ++mi) {
#pragma unroll
        for (int ni = 0; ni < 4; ++ni) {
            int col = n0 + wn + ni * 16 + nl;
            int rowb = m0 + wm + mi * 16 + hl * 8;
#pragma unroll
            for (int r = 0; r < 8; ++r)
                C[(long long)(rowb + r) * N + col] = (OutT)acc[mi][ni][r];
        }
    }
}

// ---------------- TDM 2D tile descriptor issue (128 rows x 32 bf16) ----------------
// Pads each 64B row with 16B -> LDS row stride 80B = 40 bf16 (matches LDS_STRIDE).
__device__ __forceinline__ void tdm_issue_tile(unsigned lds_off, const __bf16* gptr, int K) {
#if HAVE_TDM
    unsigned long long ga = (unsigned long long)(uintptr_t)gptr;
    u32x4 g0;
    g0.x = 1u;                                   // count=1 valid descriptor
    g0.y = lds_off;                              // lds_addr (bytes)
    g0.z = (unsigned)ga;                         // global_addr[31:0]
    g0.w = (unsigned)((ga >> 32) & 0x01FFFFFFull) | (2u << 30);  // addr[56:32] | type=2
    i32x8 g1;
    // data_size=2B(1), pad_enable, pad_interval=3 (16 DWORDs=64B), pad_amount=3 (4 DWORDs=16B)
    g1[0] = (int)((1u << 16) | (1u << 20) | (3u << 22) | (3u << 25));
    g1[1] = (int)(((unsigned)K & 0xFFFFu) << 16);               // tensor_dim0 lo16
    g1[2] = (int)(((unsigned)K >> 16) & 0xFFFFu);               // tensor_dim0 hi16 | td1 lo16=0
    g1[3] = (int)(16u | (32u << 16));                           // td1 hi16 (td1=1<<20) | tile_dim0=32
    g1[4] = (int)(128u);                                        // tile_dim1=128 | tile_dim2=0
    g1[5] = (int)K;                                             // tensor_dim0_stride lo32
    g1[6] = 0;                                                  // stride hi | dim1_stride lo
    g1[7] = 0;
    i32x4 z4 = {0, 0, 0, 0};
#if TDM_6ARG
    i32x8 z8 = {0, 0, 0, 0, 0, 0, 0, 0};
    __builtin_amdgcn_tensor_load_to_lds(g0, g1, z4, z4, z8, 0);
#else
    __builtin_amdgcn_tensor_load_to_lds(g0, g1, z4, z4, 0);
#endif
#else
    (void)lds_off; (void)gptr; (void)K;
#endif
}

// ---------------- NT GEMM (C = A * B^T), TDM double-buffered tiles ----------------
// A: M x K row-major, B: N x K row-major, C: M x N row-major (OutT), batched.
template <typename OutT>
__global__ __launch_bounds__(256) void gemm_bf16_nt_tdm_kernel(
    const __bf16* __restrict__ Aall, const __bf16* __restrict__ Ball,
    OutT* __restrict__ Call, int M, int N, int K,
    long long sA, long long sB, long long sC) {
    const int bz = blockIdx.z;
    const __bf16* A = Aall + (long long)bz * sA;
    const __bf16* Bm = Ball + (long long)bz * sB;
    OutT* C = Call + (long long)bz * sC;

    const int m0 = blockIdx.y * 128;
    const int n0 = blockIdx.x * 128;
    const int tid = threadIdx.x;
    const int lane = tid & 31;
    const int wave = tid >> 5;
    const int wm = (wave & 3) * 32;
    const int wn = (wave >> 2) * 64;

    v8f acc[2][4];
#pragma unroll
    for (int mi = 0; mi < 2; ++mi)
#pragma unroll
        for (int ni = 0; ni < 4; ++ni)
#pragma unroll
            for (int r = 0; r < 8; ++r) acc[mi][ni][r] = 0.0f;

#if HAVE_TDM
    __shared__ __bf16 As[2][128 * LDS_STRIDE];
    __shared__ __bf16 Bs[2][128 * LDS_STRIDE];
    const unsigned aOff0 = (unsigned)(uintptr_t)(&As[0][0]);
    const unsigned bOff0 = (unsigned)(uintptr_t)(&Bs[0][0]);
    const int nk = K >> 5;
    const bool issuer = (wave == 0);

    if (issuer) {
        tdm_issue_tile(aOff0, A + (long long)m0 * K, K);
        tdm_issue_tile(bOff0, Bm + (long long)n0 * K, K);
    }
    for (int i = 0; i < nk; ++i) {
        if (i + 1 < nk) {
            if (issuer) {
                const int k1 = (i + 1) << 5;
                const unsigned sel = (unsigned)((i + 1) & 1) * TILE_BYTES;
                tdm_issue_tile(aOff0 + sel, A + (long long)m0 * K + k1, K);
                tdm_issue_tile(bOff0 + sel, Bm + (long long)n0 * K + k1, K);
            }
            __builtin_amdgcn_s_wait_tensorcnt(2);  // step-i pair retired (in-order)
        } else {
            __builtin_amdgcn_s_wait_tensorcnt(0);
        }
        __syncthreads();
        mma_step(&As[i & 1][0], &Bs[i & 1][0], wm, wn, lane, acc);
        __syncthreads();  // all waves done reading before TDM overwrites this buffer
    }
#else
    // Fallback: manual staged loads (single-buffered)
    __shared__ __bf16 As[128 * LDS_STRIDE];
    __shared__ __bf16 Bs[128 * LDS_STRIDE];
    for (int k0 = 0; k0 < K; k0 += 32) {
#pragma unroll
        for (int it = 0; it < 2; ++it) {
            int c = tid + it * 256;
            int row = c >> 2;
            int col = (c & 3) * 8;
            v8bf va = *(const v8bf*)(A + (long long)(m0 + row) * K + k0 + col);
            *(v8bf*)(As + row * LDS_STRIDE + col) = va;
            v8bf vb = *(const v8bf*)(Bm + (long long)(n0 + row) * K + k0 + col);
            *(v8bf*)(Bs + row * LDS_STRIDE + col) = vb;
        }
        __syncthreads();
        mma_step(As, Bs, wm, wn, lane, acc);
        __syncthreads();
    }
#endif

    store_tile(C, N, m0, n0, wm, wn, lane, acc);
}

// ---------------- NN GEMM (C = A * B), manual loads, B transposed into LDS ----------
template <typename OutT>
__global__ __launch_bounds__(256) void gemm_bf16_nn_kernel(
    const __bf16* __restrict__ Aall, const __bf16* __restrict__ Ball,
    OutT* __restrict__ Call, int M, int N, int K,
    long long sA, long long sB, long long sC) {
    __shared__ __bf16 As[128 * LDS_STRIDE];
    __shared__ __bf16 Bs[128 * LDS_STRIDE];

    const int bz = blockIdx.z;
    const __bf16* A = Aall + (long long)bz * sA;
    const __bf16* Bm = Ball + (long long)bz * sB;
    OutT* C = Call + (long long)bz * sC;

    const int m0 = blockIdx.y * 128;
    const int n0 = blockIdx.x * 128;
    const int tid = threadIdx.x;
    const int lane = tid & 31;
    const int wave = tid >> 5;
    const int wm = (wave & 3) * 32;
    const int wn = (wave >> 2) * 64;

    v8f acc[2][4];
#pragma unroll
    for (int mi = 0; mi < 2; ++mi)
#pragma unroll
        for (int ni = 0; ni < 4; ++ni)
#pragma unroll
            for (int r = 0; r < 8; ++r) acc[mi][ni][r] = 0.0f;

    for (int k0 = 0; k0 < K; k0 += 32) {
#pragma unroll
        for (int it = 0; it < 2; ++it) {
            int c = tid + it * 256;
            int row = c >> 2;
            int col = (c & 3) * 8;
            v8bf v = *(const v8bf*)(A + (long long)(m0 + row) * K + k0 + col);
            *(v8bf*)(As + row * LDS_STRIDE + col) = v;
        }
#pragma unroll
        for (int it = 0; it < 2; ++it) {
            int c = tid + it * 256;
            int kk = c >> 4;           // 0..31
            int nc = (c & 15) * 8;     // 0..120
            v8bf v = *(const v8bf*)(Bm + (long long)(k0 + kk) * N + n0 + nc);
#pragma unroll
            for (int j = 0; j < 8; ++j) Bs[(nc + j) * LDS_STRIDE + kk] = v[j];
        }
        __syncthreads();
        mma_step(As, Bs, wm, wn, lane, acc);
        __syncthreads();
    }

    store_tile(C, N, m0, n0, wm, wn, lane, acc);
}

// ---------------- row softmax over src positions (axis -1), src mask ----------------
__global__ __launch_bounds__(256) void row_softmax_kernel(
    const float* __restrict__ sim, const int* __restrict__ smask,
    __bf16* __restrict__ ew) {
    __shared__ float red[256];
    const long long row = blockIdx.x;        // b*1024 + e
    const int b = (int)(row >> 10);
    const int t = threadIdx.x;

    float4 xv = ((const float4*)(sim + row * 1024))[t];
    int4 mv = ((const int4*)(smask + (long long)b * 1024))[t];
    float x[4] = {xv.x, xv.y, xv.z, xv.w};
    int mk[4] = {mv.x, mv.y, mv.z, mv.w};

    float m = NEG_INF;
#pragma unroll
    for (int i = 0; i < 4; ++i)
        if (!mk[i]) m = fmaxf(m, x[i]);
    red[t] = m;
    __syncthreads();
    for (int s = 128; s > 0; s >>= 1) {
        if (t < s) red[t] = fmaxf(red[t], red[t + s]);
        __syncthreads();
    }
    m = red[0];
    __syncthreads();

    float w[4];
    float acc = 0.0f;
#pragma unroll
    for (int i = 0; i < 4; ++i) {
        w[i] = mk[i] ? 0.0f : __expf(x[i] - m);
        acc += w[i];
    }
    red[t] = acc;
    __syncthreads();
    for (int s = 128; s > 0; s >>= 1) {
        if (t < s) red[t] += red[t + s];
        __syncthreads();
    }
    float inv = 1.0f / red[0];

    v4bf o;
#pragma unroll
    for (int i = 0; i < 4; ++i) o[i] = (__bf16)(w[i] * inv);
    ((v4bf*)(ew + row * 1024))[t] = o;
}

// ---------------- column softmax over edit positions (axis 1), edit mask ----------
// writes src_weights TRANSPOSED: swT[b][l][e]
__global__ __launch_bounds__(256) void col_softmax_kernel(
    const float* __restrict__ sim, const int* __restrict__ emask,
    __bf16* __restrict__ swT) {
    const int b = blockIdx.y;
    const int l = blockIdx.x * 256 + threadIdx.x;
    const float* base = sim + ((long long)b << 20);
    const int* mk = emask + (long long)b * 1024;

    float m = NEG_INF, s = 0.0f;
    for (int e = 0; e < 1024; ++e) {
        if (mk[e]) continue;                         // uniform branch per block
        float x = base[(long long)e * 1024 + l];
        float nm = fmaxf(m, x);
        s = s * __expf(m - nm) + __expf(x - nm);
        m = nm;
    }
    float inv = 1.0f / s;

    __bf16* out = swT + ((long long)b * 1024 + l) * 1024;
    for (int e = 0; e < 1024; ++e) {
        float w = mk[e] ? 0.0f
                        : __expf(base[(long long)e * 1024 + l] - m) * inv;
        out[e] = (__bf16)w;
    }
}

// ---------------- host-side orchestration ----------------
extern "C" void kernel_launch(void* const* d_in, const int* in_sizes, int n_in,
                              void* d_out, int out_size, void* d_ws, size_t ws_size,
                              hipStream_t stream) {
    (void)in_sizes; (void)n_in; (void)out_size; (void)ws_size;
    constexpr long long NB = 16, L = 1024, D = 1024;
    constexpr long long ELEM = NB * L * D;           // 16,777,216
    constexpr long long BF16_T = ELEM * 2;           // 32 MB
    constexpr long long F32_T = ELEM * 4;            // 64 MB
    constexpr long long W_BF = D * D * 2;            // 2 MB

    const float* edit = (const float*)d_in[0];
    const float* src  = (const float*)d_in[1];
    const int* emask  = (const int*)d_in[2];
    const int* smask  = (const int*)d_in[3];
    const float* W    = (const float*)d_in[4];

    char* ws = (char*)d_ws;
    __bf16* edit_bf = (__bf16*)(ws);
    __bf16* src_bf  = (__bf16*)(ws + BF16_T);
    __bf16* w_bf    = (__bf16*)(ws + 2 * BF16_T);
    __bf16* proj_bf = (__bf16*)(ws + 2 * BF16_T + W_BF);
    float*  sim     = (float*) (ws + 3 * BF16_T + W_BF);
    __bf16* ew_bf   = (__bf16*)(ws + 3 * BF16_T + W_BF + F32_T);
    __bf16* swT_bf  = (__bf16*)(ws + 4 * BF16_T + W_BF + F32_T);

    float* edit_ctx = (float*)d_out;
    float* src_ctx  = (float*)d_out + ELEM;

    // 1) convert inputs to bf16
    cvt_bf16_kernel<<<(unsigned)(ELEM / 4 / 256), 256, 0, stream>>>(edit, edit_bf, ELEM / 4);
    cvt_bf16_kernel<<<(unsigned)(ELEM / 4 / 256), 256, 0, stream>>>(src, src_bf, ELEM / 4);
    cvt_bf16_kernel<<<(unsigned)(D * D / 4 / 256), 256, 0, stream>>>(W, w_bf, D * D / 4);

    dim3 grid(8, 8, 16), blk(256);
    // 2) proj = edit @ W^T  (NT, TDM tiles; W batch-broadcast) -> bf16
    gemm_bf16_nt_tdm_kernel<__bf16><<<grid, blk, 0, stream>>>(
        edit_bf, w_bf, proj_bf, 1024, 1024, 1024, L * D, 0, L * D);
    // 3) sim = proj @ src^T (NT, TDM tiles) -> f32
    gemm_bf16_nt_tdm_kernel<float><<<grid, blk, 0, stream>>>(
        proj_bf, src_bf, sim, 1024, 1024, 1024, L * D, L * D, L * L);
    // 4) softmax over src positions (mask = src) -> edit_weights (bf16)
    row_softmax_kernel<<<(unsigned)(NB * L), 256, 0, stream>>>(sim, smask, ew_bf);
    // 5) softmax over edit positions (mask = edit) -> src_weights^T (bf16)
    col_softmax_kernel<<<dim3(4, 16), 256, 0, stream>>>(sim, emask, swT_bf);
    // 6) edit_ctx = edit_weights @ src (NN) -> f32 output
    gemm_bf16_nn_kernel<float><<<grid, blk, 0, stream>>>(
        ew_bf, src_bf, edit_ctx, 1024, 1024, 1024, L * L, L * D, L * D);
    // 7) src_ctx = sw^T @ edit (NN) -> f32 output
    gemm_bf16_nn_kernel<float><<<grid, blk, 0, stream>>>(
        swT_bf, edit_bf, src_ctx, 1024, 1024, 1024, L * L, L * D, L * D);
}
